// Utango_31791347925838
// MI455X (gfx1250) — matrix operand order repaired and checked by reading, not compiled, not run on hardware
//
#include <hip/hip_runtime.h>

// ---------------------------------------------------------------------------
// GCN 7-branch pipeline for MI455X (gfx1250, wave32, WMMA).
// Dense GEMMs use split-bf16 v_wmma_f32_16x16x32_bf16 (3 WMMAs per 32-K block
// gives ~fp32 accuracy). Edge scatter uses one wave per edge with coalesced
// float4 loads + global f32 atomics (working set is L2-resident: 192 MB L2).
// ---------------------------------------------------------------------------

typedef unsigned short u16;
typedef unsigned int   u32;
typedef __attribute__((ext_vector_type(4)))  u32    v4u;
typedef __attribute__((ext_vector_type(16))) __bf16 v16bf;
typedef __attribute__((ext_vector_type(8)))  float  v8f;

#define HDIM 128

__device__ __forceinline__ u16 f2bf(float f) {            // fp32 -> bf16 RNE
  u32 u = __float_as_uint(f);
  u += 0x7FFFu + ((u >> 16) & 1u);
  return (u16)(u >> 16);
}
__device__ __forceinline__ float bf2f(u16 h) {
  return __uint_as_float(((u32)h) << 16);
}

union Frag32B { v4u q[2]; v16bf v; };

// two independent 16B chunks -> one 32B bf16 fragment (8 VGPRs)
__device__ __forceinline__ v16bf load_frag2(const u16* p0, const u16* p1) {
  Frag32B f;
  f.q[0] = *(const v4u*)p0;
  f.q[1] = *(const v4u*)p1;
  return f.v;
}

// ----------------------------- utility kernels -----------------------------

__global__ void k_fill_f32(float* __restrict__ p, float v, long n) {
  long i = (long)blockIdx.x * blockDim.x + threadIdx.x;
  if (i < n) p[i] = v;
}

__global__ void k_edge_deg(const int* __restrict__ dst, float* __restrict__ deg, int E) {
  int i = blockIdx.x * blockDim.x + threadIdx.x;
  if (i < E) atomicAdd(&deg[dst[i]], 1.0f);
}

// deg (in d_invdeg) -> dinv_sqrt, invdeg in place
__global__ void k_node_norm(float* __restrict__ deg_io, float* __restrict__ dinv, int N) {
  int i = blockIdx.x * blockDim.x + threadIdx.x;
  if (i < N) {
    float d = deg_io[i];
    dinv[i]   = 1.0f / sqrtf(d);
    deg_io[i] = 1.0f / d;
  }
}

__global__ void k_edge_norm(const int* __restrict__ src, const int* __restrict__ dst,
                            const float* __restrict__ dinv, float* __restrict__ norm, int E) {
  int i = blockIdx.x * blockDim.x + threadIdx.x;
  if (i < E) norm[i] = dinv[src[i]] * dinv[dst[i]];
}

// fp32 -> (bf16 hi, bf16 lo) split
__global__ void k_split(const float* __restrict__ s, u16* __restrict__ hi,
                        u16* __restrict__ lo, long n) {
  long i = (long)blockIdx.x * blockDim.x + threadIdx.x;
  if (i >= n) return;
  float f = s[i];
  u16 h = f2bf(f);
  hi[i] = h;
  lo[i] = f2bf(f - bf2f(h));
}

// W[k][n] (row-major 128x128) -> transposed split Wt_hi/Wt_lo[n][k]
__global__ void k_wt_split(const float* __restrict__ W, u16* __restrict__ wthi,
                           u16* __restrict__ wtlo) {
  int t = blockIdx.x * blockDim.x + threadIdx.x;   // 0..16383
  int k = t >> 7, n = t & 127;
  float f = W[t];                                   // W[k][n]
  u16 h = f2bf(f);
  int o = n * HDIM + k;
  wthi[o] = h;
  wtlo[o] = f2bf(f - bf2f(h));
}

// ------------------------------- WMMA GEMM ---------------------------------
// C[N,128] = A[N,128] x W[128,128], A given as bf16 hi/lo, W as transposed
// bf16 hi/lo (Wt[n][k]). One block = 16 rows; wave w -> 16-col tile w.
__global__ __launch_bounds__(256) void k_gemm_split(
    const u16* __restrict__ Ahi, const u16* __restrict__ Alo,
    const u16* __restrict__ Bhi, const u16* __restrict__ Blo,
    float* __restrict__ C, int nrows) {
  const int lane = threadIdx.x & 31;
  const int wave = threadIdx.x >> 5;     // column tile 0..7
  const int row0 = blockIdx.x << 4;
  const int col0 = wave << 4;
  const int m = lane & 15;
  const int h = lane >> 4;

  int rA = row0 + m;
  if (rA >= nrows) rA = nrows - 1;       // clamp: keep EXEC all-ones for WMMA
  const u16* arow_hi = Ahi + (size_t)rA * HDIM;
  const u16* arow_lo = Alo + (size_t)rA * HDIM;
  const u16* brow_hi = Bhi + (size_t)(col0 + m) * HDIM;
  const u16* brow_lo = Blo + (size_t)(col0 + m) * HDIM;

  v8f acc = {};
#pragma unroll
  for (int kb = 0; kb < HDIM; kb += 32) {
    // A frag: lane half h holds K = kb+16h .. +16 of row m (matches B)
    int ka0 = kb + 16 * h, ka1 = kb + 16 * h + 8;
    v16bf a_hi = load_frag2(arow_hi + ka0, arow_hi + ka1);
    v16bf a_lo = load_frag2(arow_lo + ka0, arow_lo + ka1);
    // B frag (contiguous k order, same as A): lane half h holds K = kb+16h .. +16 of col n
    int kbb = kb + 16 * h;
    v16bf b_hi = load_frag2(brow_hi + kbb, brow_hi + kbb + 8);
    v16bf b_lo = load_frag2(brow_lo + kbb, brow_lo + kbb + 8);
    // split product: (ah+al)(bh+bl) ~= ah*bh + al*bh + ah*bl
    acc = __builtin_amdgcn_wmma_f32_16x16x32_bf16(false, a_hi, false, b_hi, (short)0, acc, false, false);
    acc = __builtin_amdgcn_wmma_f32_16x16x32_bf16(false, a_lo, false, b_hi, (short)0, acc, false, false);
    acc = __builtin_amdgcn_wmma_f32_16x16x32_bf16(false, a_hi, false, b_lo, (short)0, acc, false, false);
  }
  // C/D f32 16x16 layout: VGPR r -> (M = r + 8h, N = lane&15)
#pragma unroll
  for (int r = 0; r < 8; r++) {
    int row = row0 + r + 8 * h;
    if (row < nrows) C[(size_t)row * HDIM + col0 + m] = acc[r];
  }
}

// --------------------------- edge scatter-add ------------------------------
// One wave per edge: lane l handles floats [4l,4l+4). Coalesced 512B/wave
// gather from xw[src], 4 coalesced f32 atomics/lane into agg[dst].
__global__ void k_edge_agg(const int* __restrict__ src, const int* __restrict__ dst,
                           const float* __restrict__ xw, const float* __restrict__ norm,
                           float* __restrict__ agg, int E) {
  int gt = blockIdx.x * blockDim.x + threadIdx.x;
  int e = gt >> 5, lane = gt & 31;
  if (e >= E) return;
  int s = src[e], d = dst[e];
  float w = norm[e];
  float4 v = ((const float4*)(xw + (size_t)s * HDIM))[lane];
  float* p = agg + (size_t)d * HDIM + lane * 4;
  atomicAdd(p + 0, v.x * w);
  atomicAdd(p + 1, v.y * w);
  atomicAdd(p + 2, v.z * w);
  atomicAdd(p + 3, v.w * w);
}

// h1 = relu(agg + xw*invdeg + bias) -> split bf16 hi/lo (GEMM2 input)
__global__ void k_combine_relu_split(const float* __restrict__ agg, const float* __restrict__ xw,
                                     const float* __restrict__ invdeg, const float* __restrict__ bias,
                                     u16* __restrict__ hi, u16* __restrict__ lo, long n) {
  long i = (long)blockIdx.x * blockDim.x + threadIdx.x;
  if (i >= n) return;
  int node = (int)(i >> 7), c = (int)(i & 127);
  float v = agg[i] + xw[i] * invdeg[node] + bias[c];
  v = fmaxf(v, 0.0f);
  u16 h = f2bf(v);
  hi[i] = h;
  lo[i] = f2bf(v - bf2f(h));
}

// Final: fv = agg + xw*invdeg + bias; logits = fv @ hw^T + hb; softmax.
// One wave per node; wave32 __shfl_xor reduction.
__global__ void k_head(const float* __restrict__ agg, const float* __restrict__ xw,
                       const float* __restrict__ invdeg, const float* __restrict__ bias,
                       const float* __restrict__ hw, const float* __restrict__ hb,
                       float* __restrict__ out, int nc, int N) {
  int gt = blockIdx.x * blockDim.x + threadIdx.x;
  int node = gt >> 5, lane = gt & 31;
  if (node >= N) return;
  float id = invdeg[node];
  float4 a  = ((const float4*)(agg + (size_t)node * HDIM))[lane];
  float4 xv = ((const float4*)(xw  + (size_t)node * HDIM))[lane];
  float4 b  = ((const float4*)bias)[lane];
  float4 fv;
  fv.x = a.x + xv.x * id + b.x;
  fv.y = a.y + xv.y * id + b.y;
  fv.z = a.z + xv.z * id + b.z;
  fv.w = a.w + xv.w * id + b.w;
  float lg[3];
  for (int j = 0; j < nc; j++) {
    float4 w = ((const float4*)(hw + j * HDIM))[lane];
    float p = fv.x * w.x + fv.y * w.y + fv.z * w.z + fv.w * w.w;
    for (int o = 16; o > 0; o >>= 1) p += __shfl_xor(p, o, 32);
    lg[j] = p + hb[j];
  }
  float mx = lg[0];
  for (int j = 1; j < nc; j++) mx = fmaxf(mx, lg[j]);
  float s = 0.f, ex[3];
  for (int j = 0; j < nc; j++) { ex[j] = expf(lg[j] - mx); s += ex[j]; }
  if (lane < nc) out[(size_t)node * nc + lane] = ex[lane] / s;
}

// ------------------------------- launcher ----------------------------------

extern "C" void kernel_launch(void* const* d_in, const int* in_sizes, int n_in,
                              void* d_out, int out_size, void* d_ws, size_t ws_size,
                              hipStream_t stream) {
  const float* x      = (const float*)d_in[0];
  const int*   edges  = (const int*)d_in[1];   // int32 per harness convention
  const float* gcn_w  = (const float*)d_in[2];
  const float* gcn_b  = (const float*)d_in[3];
  const float* rs3_w  = (const float*)d_in[4];
  const float* rs3_b  = (const float*)d_in[5];
  const float* rs2_w  = (const float*)d_in[6];
  const float* rs2_b  = (const float*)d_in[7];
  (void)n_in; (void)out_size; (void)ws_size;

  const long N = in_sizes[0] / HDIM;
  const long E = in_sizes[1] / 2;
  const int* src = edges;
  const int* dst = edges + E;

  // --- workspace carve-out (256B aligned) ---
  char* ws = (char*)d_ws;
  auto carve = [&](size_t bytes) {
    char* p = ws;
    ws += (bytes + 255) & ~(size_t)255;
    return p;
  };
  float* d_invdeg = (float*)carve((size_t)N * 4);            // deg -> 1/deg
  float* d_dinv   = (float*)carve((size_t)N * 4);            // deg^-0.5
  float* d_norm   = (float*)carve((size_t)E * 4);            // per-edge norm
  float* d_xw     = (float*)carve((size_t)N * HDIM * 4);     // xw / xw2
  float* d_agg    = (float*)carve((size_t)N * HDIM * 4);     // agg / agg2
  u16*   d_ahi    = (u16*)carve((size_t)N * HDIM * 2);       // GEMM A hi
  u16*   d_alo    = (u16*)carve((size_t)N * HDIM * 2);       // GEMM A lo
  u16*   d_wthi   = (u16*)carve((size_t)HDIM * HDIM * 2);    // Wt hi
  u16*   d_wtlo   = (u16*)carve((size_t)HDIM * HDIM * 2);    // Wt lo

  const int T = 256;
  auto nb = [](long n, int t) { return (unsigned)((n + t - 1) / t); };
  const long NH = N * HDIM;

  // --- graph normalization (once) ---
  k_fill_f32<<<nb(N, T), T, 0, stream>>>(d_invdeg, 1.0f, N);       // self-loop
  k_edge_deg<<<nb(E, T), T, 0, stream>>>(dst, d_invdeg, (int)E);
  k_node_norm<<<nb(N, T), T, 0, stream>>>(d_invdeg, d_dinv, (int)N);
  k_edge_norm<<<nb(E, T), T, 0, stream>>>(src, dst, d_dinv, d_norm, (int)E);

  const unsigned gGemm = nb(N, 16);
  const unsigned gEdge = nb(E * 32, T);
  const unsigned gNode = nb(N * 32, T);
  const unsigned gElem = nb(NH, T);

  for (int k = 0; k < 7; k++) {
    const float* W = gcn_w + (size_t)k * HDIM * HDIM;
    const float* b = gcn_b + (size_t)k * HDIM;

    // conv1
    k_split<<<gElem, T, 0, stream>>>(x, d_ahi, d_alo, NH);
    k_wt_split<<<64, T, 0, stream>>>(W, d_wthi, d_wtlo);
    k_gemm_split<<<gGemm, T, 0, stream>>>(d_ahi, d_alo, d_wthi, d_wtlo, d_xw, (int)N);
    k_fill_f32<<<gElem, T, 0, stream>>>(d_agg, 0.0f, NH);
    k_edge_agg<<<gEdge, T, 0, stream>>>(src, dst, d_xw, d_norm, d_agg, (int)E);
    k_combine_relu_split<<<gElem, T, 0, stream>>>(d_agg, d_xw, d_invdeg, b, d_ahi, d_alo, NH);

    // conv2 (same W)
    k_gemm_split<<<gGemm, T, 0, stream>>>(d_ahi, d_alo, d_wthi, d_wtlo, d_xw, (int)N);
    k_fill_f32<<<gElem, T, 0, stream>>>(d_agg, 0.0f, NH);
    k_edge_agg<<<gEdge, T, 0, stream>>>(src, dst, d_xw, d_norm, d_agg, (int)E);

    // head + softmax
    const float* hw; const float* hb; float* outp; int nc;
    if (k < 5) {
      nc = 3; hw = rs3_w + (size_t)k * 3 * HDIM; hb = rs3_b + (size_t)k * 3;
      outp = (float*)d_out + (size_t)k * N * 3;
    } else {
      nc = 2; hw = rs2_w + (size_t)(k - 5) * 2 * HDIM; hb = rs2_b + (size_t)(k - 5) * 2;
      outp = (float*)d_out + (size_t)5 * N * 3 + (size_t)(k - 5) * N * 2;
    }
    k_head<<<gNode, T, 0, stream>>>(d_agg, d_xw, d_invdeg, b, hw, hb, outp, nc, (int)N);
  }
}
